// PartSegmentation_36352603193850
// MI455X (gfx1250) — matrix operand order repaired and checked by reading, not compiled, not run on hardware
//
#include <hip/hip_runtime.h>
#include <hip/hip_bf16.h>

// ---------------------------------------------------------------------------
// PointNet++ part segmentation forward for gfx1250 (MI455X).
// All per-point MLPs run as WMMA f16 GEMMs (v_wmma_f32_16x16x32_f16),
// f32 accumulate, operands loaded directly in the documented CDNA5 WMMA
// VGPR layout (no LDS staging). FPS is a persistent-block sequential kernel.
// ---------------------------------------------------------------------------

typedef __attribute__((ext_vector_type(16))) _Float16 v16h;
typedef __attribute__((ext_vector_type(8)))  float    v8f;
typedef __attribute__((ext_vector_type(4)))  float    f4v;

union AV { v16h h; f4v f[2]; };

#define DEVFN static __device__ __forceinline__

// ---------------- transpose (B,3,N) -> (B,N,3), B=16, N=4096 ----------------
__global__ void k_transpose(const float* __restrict__ in, float* __restrict__ pc) {
  int i = blockIdx.x * 256 + threadIdx.x;        // 0..B*N-1
  int n = i & 4095, b = i >> 12;
  const float* s = in + (size_t)b * 3 * 4096 + n;
  float* d = pc + (size_t)i * 3;
  d[0] = s[0]; d[1] = s[4096]; d[2] = s[2 * 4096];
}

// ---------------- farthest point sampling: one block per batch --------------
template <int NPTS>
__global__ __launch_bounds__(256) void k_fps(const float* __restrict__ xyz,
                                             int* __restrict__ out, int npoint) {
  constexpr int PPT = NPTS / 256;
  __shared__ float sv[256];
  __shared__ int   si[256];
  int b = blockIdx.x, t = threadIdx.x;
  const float* base = xyz + (size_t)b * NPTS * 3;
  float px[PPT], py[PPT], pz[PPT], dist[PPT];
  for (int j = 0; j < PPT; ++j) {
    int p = j * 256 + t;
    px[j] = base[p * 3 + 0]; py[j] = base[p * 3 + 1]; pz[j] = base[p * 3 + 2];
    dist[j] = 1e10f;
  }
  int far = 0;
  for (int it = 0; it < npoint; ++it) {
    if (t == 0) out[b * npoint + it] = far;
    float cx = base[far * 3 + 0], cy = base[far * 3 + 1], cz = base[far * 3 + 2];
    float bv = -1.0f; int bi = 0x7fffffff;
    for (int j = 0; j < PPT; ++j) {
      float dx = px[j] - cx, dy = py[j] - cy, dz = pz[j] - cz;
      float dd = dx * dx + dy * dy + dz * dz;
      dist[j] = fminf(dist[j], dd);
      int p = j * 256 + t;
      if (dist[j] > bv) { bv = dist[j]; bi = p; }
    }
    sv[t] = bv; si[t] = bi;
    __syncthreads();
    for (int s = 128; s > 0; s >>= 1) {
      if (t < s) {
        float ov = sv[t + s]; int oi = si[t + s];
        if (ov > sv[t] || (ov == sv[t] && oi < si[t])) { sv[t] = ov; si[t] = oi; }
      }
      __syncthreads();
    }
    far = si[0];
    __syncthreads();
  }
}

// ---------------- gather 3-vectors by index ---------------------------------
__global__ void k_gather3(const float* __restrict__ src, int Nsrc, int S,
                          const int* __restrict__ idx, float* __restrict__ dst,
                          int total) {
  int i = blockIdx.x * 256 + threadIdx.x;
  if (i >= total) return;
  int b = i / S;
  int id = idx[i];
  const float* s = src + ((size_t)b * Nsrc + id) * 3;
  float* d = dst + (size_t)i * 3;
  d[0] = s[0]; d[1] = s[1]; d[2] = s[2];
}

// ---------------- ball query: one wave32 per query, ordered append ----------
__global__ void k_ball_query(const float* __restrict__ xyz,
                             const float* __restrict__ nxyz,
                             int* __restrict__ out,
                             int S, int nsample, float r2, int N) {
  int gid = blockIdx.x * 256 + threadIdx.x;
  int lane = threadIdx.x & 31;
  int q = gid >> 5;                               // b*S + s
  int b = q / S;
  const float* base = xyz + (size_t)b * N * 3;
  float qx = nxyz[q * 3 + 0], qy = nxyz[q * 3 + 1], qz = nxyz[q * 3 + 2];
  int* o = out + (size_t)q * nsample;
  int count = 0, first = 0;
  for (int s0 = 0; s0 < N && count < nsample; s0 += 32) {
    int p = s0 + lane;
    float dx = base[p * 3 + 0] - qx, dy = base[p * 3 + 1] - qy, dz = base[p * 3 + 2] - qz;
    float dd = dx * dx + dy * dy + dz * dz;
    bool in = dd <= r2;
    unsigned mask = __builtin_amdgcn_ballot_w32(in);
    if (count == 0 && mask) first = s0 + __ffs(mask) - 1;
    int pos = count + __popc(mask & ((1u << lane) - 1u));
    if (in && pos < nsample) o[pos] = p;
    count += __popc(mask);
  }
  for (int j = count + lane; j < nsample; j += 32) o[j] = first;
}

// ---------------- grouping kernels (emit padded f16 GEMM inputs) ------------
// SA1: rows = B*512*32, K=6 padded to 32. ch0-2 rel xyz, ch3-5 abs xyz.
__global__ void k_group_sa1(const float* __restrict__ pc, const float* __restrict__ nxyz,
                            const int* __restrict__ idx, _Float16* __restrict__ X) {
  int row = blockIdx.x * 256 + threadIdx.x;       // (b*512+s)*32 + j
  int bs = row >> 5;
  int id = idx[row];
  int b = bs >> 9;
  const float* p = pc + ((size_t)b * 4096 + id) * 3;
  const float* q = nxyz + (size_t)bs * 3;
  _Float16* o = X + (size_t)row * 32;
  o[0] = (_Float16)(p[0] - q[0]); o[1] = (_Float16)(p[1] - q[1]); o[2] = (_Float16)(p[2] - q[2]);
  o[3] = (_Float16)p[0]; o[4] = (_Float16)p[1]; o[5] = (_Float16)p[2];
  for (int c = 6; c < 32; ++c) o[c] = (_Float16)0.f;
}

// SA2: rows = B*128*64, K=131 padded to 160. ch0-2 rel xyz, ch3-130 f1[id].
__global__ void k_group_sa2(const float* __restrict__ xyz1, const float* __restrict__ nxyz,
                            const int* __restrict__ idx, const _Float16* __restrict__ f1,
                            _Float16* __restrict__ X) {
  int row = blockIdx.x * 256 + threadIdx.x;       // (b*128+s)*64 + j
  int bs = row >> 6;
  int id = idx[row];
  int b = bs >> 7;
  const float* p = xyz1 + ((size_t)b * 512 + id) * 3;
  const float* q = nxyz + (size_t)bs * 3;
  const _Float16* f = f1 + ((size_t)b * 512 + id) * 128;
  _Float16* o = X + (size_t)row * 160;
  o[0] = (_Float16)(p[0] - q[0]); o[1] = (_Float16)(p[1] - q[1]); o[2] = (_Float16)(p[2] - q[2]);
  for (int c = 0; c < 128; ++c) o[3 + c] = f[c];
  for (int c = 131; c < 160; ++c) o[c] = (_Float16)0.f;
}

// SA3 global: rows = B*128, K=259 padded to 288. ch0-2 abs xyz2, ch3-258 f2.
__global__ void k_build_x3(const float* __restrict__ xyz2, const _Float16* __restrict__ f2,
                           _Float16* __restrict__ X) {
  int row = blockIdx.x * 256 + threadIdx.x;       // 2048 rows
  _Float16* o = X + (size_t)row * 288;
  o[0] = (_Float16)xyz2[row * 3 + 0];
  o[1] = (_Float16)xyz2[row * 3 + 1];
  o[2] = (_Float16)xyz2[row * 3 + 2];
  const _Float16* f = f2 + (size_t)row * 256;
  for (int c = 0; c < 256; ++c) o[3 + c] = f[c];
  for (int c = 259; c < 288; ++c) o[c] = (_Float16)0.f;
}

// FP1: rows = B*128, K=1280 = concat(f2[256], broadcast f3[1024]).
__global__ void k_build_xf1(const _Float16* __restrict__ f2, const float* __restrict__ f3,
                            _Float16* __restrict__ X) {
  int row = blockIdx.x * 256 + threadIdx.x;       // 2048 rows
  int b = row >> 7;
  _Float16* o = X + (size_t)row * 1280;
  const _Float16* f = f2 + (size_t)row * 256;
  for (int c = 0; c < 256; ++c) o[c] = f[c];
  const float* g = f3 + (size_t)b * 1024;
  for (int c = 0; c < 1024; ++c) o[256 + c] = (_Float16)g[c];
}

// ---------------- 3-NN with inverse-distance weights ------------------------
DEVFN void knn3(float qx, float qy, float qz, const float* __restrict__ src, int ns,
                int& i0, int& i1, int& i2, float& w0, float& w1, float& w2) {
  float d0 = 1e30f, d1 = 1e30f, d2 = 1e30f;
  i0 = i1 = i2 = 0;
  for (int m = 0; m < ns; ++m) {
    float dx = src[m * 3 + 0] - qx, dy = src[m * 3 + 1] - qy, dz = src[m * 3 + 2] - qz;
    float dd = dx * dx + dy * dy + dz * dz;
    if (dd < d0)      { d2 = d1; i2 = i1; d1 = d0; i1 = i0; d0 = dd; i0 = m; }
    else if (dd < d1) { d2 = d1; i2 = i1; d1 = dd; i1 = m; }
    else if (dd < d2) { d2 = dd; i2 = m; }
  }
  w0 = 1.f / (d0 + 1e-8f); w1 = 1.f / (d1 + 1e-8f); w2 = 1.f / (d2 + 1e-8f);
  float s = w0 + w1 + w2;
  w0 /= s; w1 /= s; w2 /= s;
}

// FP2: rows = B*512, K=384 = concat(f1[128], interp3(f4)[256]).
__global__ void k_fp2_build(const float* __restrict__ xyz1, const float* __restrict__ xyz2,
                            const _Float16* __restrict__ f1, const _Float16* __restrict__ f4,
                            _Float16* __restrict__ X) {
  int row = blockIdx.x * 256 + threadIdx.x;       // 8192 rows = b*512+s
  int b = row >> 9;
  float qx = xyz1[row * 3 + 0], qy = xyz1[row * 3 + 1], qz = xyz1[row * 3 + 2];
  int i0, i1, i2; float w0, w1, w2;
  knn3(qx, qy, qz, xyz2 + (size_t)b * 128 * 3, 128, i0, i1, i2, w0, w1, w2);
  _Float16* o = X + (size_t)row * 384;
  const _Float16* fr = f1 + (size_t)row * 128;
  for (int c = 0; c < 128; ++c) o[c] = fr[c];
  const _Float16* a = f4 + ((size_t)b * 128 + i0) * 256;
  const _Float16* bb = f4 + ((size_t)b * 128 + i1) * 256;
  const _Float16* cc = f4 + ((size_t)b * 128 + i2) * 256;
  for (int c = 0; c < 256; ++c)
    o[128 + c] = (_Float16)(w0 * (float)a[c] + w1 * (float)bb[c] + w2 * (float)cc[c]);
}

// FP3: rows = B*4096, K=134 padded 160 = concat([pc,pc][6], interp3(f5)[128]).
__global__ void k_fp3_build(const float* __restrict__ pc, const float* __restrict__ xyz1,
                            const _Float16* __restrict__ f5, _Float16* __restrict__ X) {
  int row = blockIdx.x * 256 + threadIdx.x;       // 65536 rows = b*4096+n
  int b = row >> 12;
  float qx = pc[row * 3 + 0], qy = pc[row * 3 + 1], qz = pc[row * 3 + 2];
  int i0, i1, i2; float w0, w1, w2;
  knn3(qx, qy, qz, xyz1 + (size_t)b * 512 * 3, 512, i0, i1, i2, w0, w1, w2);
  _Float16* o = X + (size_t)row * 160;
  o[0] = (_Float16)qx; o[1] = (_Float16)qy; o[2] = (_Float16)qz;
  o[3] = (_Float16)qx; o[4] = (_Float16)qy; o[5] = (_Float16)qz;
  const _Float16* a = f5 + ((size_t)b * 512 + i0) * 128;
  const _Float16* bb = f5 + ((size_t)b * 512 + i1) * 128;
  const _Float16* cc = f5 + ((size_t)b * 512 + i2) * 128;
  for (int c = 0; c < 128; ++c)
    o[6 + c] = (_Float16)(w0 * (float)a[c] + w1 * (float)bb[c] + w2 * (float)cc[c]);
  for (int c = 134; c < 160; ++c) o[c] = (_Float16)0.f;
}

// ---------------- weight f32(K,C) -> f16 N-major (CP,KP), zero padded -------
__global__ void k_convw(const float* __restrict__ W, _Float16* __restrict__ Wt,
                        int K, int C, int KP, int CP) {
  int i = blockIdx.x * 256 + threadIdx.x;
  if (i >= CP * KP) return;
  int k = i % KP, n = i / KP;
  Wt[i] = (n < C && k < K) ? (_Float16)W[(size_t)k * C + n] : (_Float16)0.f;
}

__global__ void k_padvec(const float* __restrict__ v, float* __restrict__ o, int C, int CP) {
  int i = blockIdx.x * 256 + threadIdx.x;
  if (i < CP) o[i] = (i < C) ? v[i] : 0.f;
}

// ---------------- WMMA GEMM: Y = epilogue(X[M,KP] @ Wt[CP,KP]^T) ------------
// One wave per 16x16 tile, 8 waves (128 rows) per block.
// flags: bit0 = relu, bit1 = has (g,be) scale.
__global__ __launch_bounds__(256) void k_gemm(
    const _Float16* __restrict__ X, int ldx,
    const _Float16* __restrict__ Wt, int ldw,
    const float* __restrict__ bias, const float* __restrict__ gg,
    const float* __restrict__ bb,
    _Float16* __restrict__ Y, int ldy,
    float* __restrict__ Yf, int ldyf,
    int ktiles, int flags) {
  int wave = threadIdx.x >> 5;
  int lane = threadIdx.x & 31;
  int rowTile = blockIdx.x * 128 + wave * 16;
  int colTile = blockIdx.y * 16;
  int r = rowTile + (lane & 15);
  int n = colTile + (lane & 15);
  int grp = lane >> 4;
  const _Float16* xrow = X + (size_t)r * ldx + grp * 8;     // A: K chunks grp*8, 16+grp*8
  const _Float16* wrow = Wt + (size_t)n * ldw + grp * 8;    // B: K chunks grp*8, 16+grp*8
  v8f acc = {};
  for (int kt = 0; kt < ktiles; ++kt) {
    AV a, bm;
    a.f[0]  = *(const f4v*)(xrow + kt * 32);
    a.f[1]  = *(const f4v*)(xrow + kt * 32 + 16);
    bm.f[0] = *(const f4v*)(wrow + kt * 32);
    bm.f[1] = *(const f4v*)(wrow + kt * 32 + 16);
    acc = __builtin_amdgcn_wmma_f32_16x16x32_f16(false, a.h, false, bm.h,
                                                 (short)0, acc, false, false);
  }
  float bv = bias ? bias[n] : 0.f;
  float gv = (flags & 2) ? gg[n] : 1.f;
  float bev = (flags & 2) ? bb[n] : 0.f;
#pragma unroll
  for (int v = 0; v < 8; ++v) {
    int row = rowTile + grp * 8 + v;                        // C/D layout: M = v + grp*8
    float val = acc[v] + bv;
    if (flags & 2) val = val * gv + bev;
    if (flags & 1) val = val > 0.f ? val : 0.f;
    if (Y)  Y[(size_t)row * ldy + n] = (_Float16)val;
    if (Yf) Yf[(size_t)row * ldyf + n] = val;
  }
}

// ---------------- max over group of rows ------------------------------------
__global__ void k_maxpool(const _Float16* __restrict__ Y, int ld, int gs,
                          _Float16* __restrict__ o16, float* __restrict__ o32,
                          int C, int total) {
  int i = blockIdx.x * 256 + threadIdx.x;
  if (i >= total) return;
  int c = i % C; int g = i / C;
  const _Float16* p = Y + (size_t)g * gs * ld + c;
  float m = -1e30f;
  for (int j = 0; j < gs; ++j) {
    float v = (float)p[(size_t)j * ld];
    m = v > m ? v : m;
  }
  if (o16) o16[i] = (_Float16)m;
  if (o32) o32[i] = m;
}

// ---------------- log_softmax over 50 classes + transpose to (B,50,N) -------
__global__ void k_logsm(const float* __restrict__ logits, float* __restrict__ out) {
  int row = blockIdx.x * 256 + threadIdx.x;       // 65536 = b*4096+n
  int n = row & 4095, b = row >> 12;
  const float* x = logits + (size_t)row * 64;
  float m = -1e30f;
  for (int c = 0; c < 50; ++c) m = fmaxf(m, x[c]);
  float s = 0.f;
  for (int c = 0; c < 50; ++c) s += expf(x[c] - m);
  float lse = m + logf(s);
  float* o = out + (size_t)b * 50 * 4096 + n;
  for (int c = 0; c < 50; ++c) o[(size_t)c * 4096] = x[c] - lse;
}

__global__ void k_copyf(const float* __restrict__ s, float* __restrict__ d, int n) {
  int i = blockIdx.x * 256 + threadIdx.x;
  if (i < n) d[i] = s[i];
}

// ---------------------------------------------------------------------------
struct Lyr { int wi, K, C, KP, CP; };

extern "C" void kernel_launch(void* const* d_in, const int* in_sizes, int n_in,
                              void* d_out, int out_size, void* d_ws, size_t ws_size,
                              hipStream_t stream) {
  (void)in_sizes; (void)n_in; (void)out_size; (void)ws_size;
  char* wsp = (char*)d_ws;
  size_t off = 0;
  auto alloc = [&](size_t bytes) -> void* {
    off = (off + 255) & ~(size_t)255;
    void* p = wsp + off;
    off += bytes;
    return p;
  };

  // persistent buffers
  float*     pc   = (float*)alloc((size_t)16 * 4096 * 3 * 4);
  int*       fps1 = (int*)alloc((size_t)16 * 512 * 4);
  float*     xyz1 = (float*)alloc((size_t)16 * 512 * 3 * 4);
  int*       bq1  = (int*)alloc((size_t)16 * 512 * 32 * 4);
  _Float16*  f1   = (_Float16*)alloc((size_t)16 * 512 * 128 * 2);
  int*       fps2 = (int*)alloc((size_t)16 * 128 * 4);
  float*     xyz2 = (float*)alloc((size_t)16 * 128 * 3 * 4);
  int*       bq2  = (int*)alloc((size_t)16 * 128 * 64 * 4);
  _Float16*  f2   = (_Float16*)alloc((size_t)16 * 128 * 256 * 2);
  float*     f3f  = (float*)alloc((size_t)16 * 1024 * 4);
  _Float16*  f4   = (_Float16*)alloc((size_t)16 * 128 * 256 * 2);
  _Float16*  f5   = (_Float16*)alloc((size_t)16 * 512 * 128 * 2);
  float*     b2p  = (float*)alloc(64 * 4);
  // ping-pong activation buffers (sized to SA1/SA2 peaks)
  _Float16*  G0   = (_Float16*)alloc((size_t)131072 * 160 * 2);  // 41.9 MB
  _Float16*  G1   = (_Float16*)alloc((size_t)262144 * 128 * 2);  // 67.1 MB

  // layer table: {weight d_in index, K, Cout, Kpad, CoutPad}
  static const Lyr L[18] = {
    {1,    6,   64,   32,   64}, {5,  64,   64,  64,   64}, {9,  64,  128,  64,  128},
    {13, 131,  128,  160,  128}, {17, 128, 128, 128,  128}, {21, 128, 256, 128,  256},
    {25, 259,  256,  288,  256}, {29, 256, 512, 256,  512}, {33, 512, 1024, 512, 1024},
    {37, 1280, 256, 1280,  256}, {41, 256, 256, 256,  256},
    {45, 384,  256,  384,  256}, {49, 256, 128, 256,  128},
    {53, 134,  128,  160,  128}, {57, 128, 128, 128,  128}, {61, 128, 128, 128, 128},
    {65, 128,  128,  128,  128}, {69, 128,  50, 128,   64},
  };
  _Float16* Wt[18];
  for (int i = 0; i < 18; ++i) {
    int sz = L[i].KP * L[i].CP;
    Wt[i] = (_Float16*)alloc((size_t)sz * 2);
    k_convw<<<(sz + 255) / 256, 256, 0, stream>>>(
        (const float*)d_in[L[i].wi], Wt[i], L[i].K, L[i].C, L[i].KP, L[i].CP);
  }
  k_padvec<<<1, 64, 0, stream>>>((const float*)d_in[70], b2p, 50, 64);

  auto gemm = [&](const void* X, int ldx, int li,
                  const void* b_, const void* g_, const void* be_,
                  void* Y, int ldy, void* Yf, int ldyf, int M, int flags) {
    dim3 grid(M / 128, L[li].CP / 16);
    k_gemm<<<grid, 256, 0, stream>>>((const _Float16*)X, ldx, Wt[li], L[li].KP,
                                     (const float*)b_, (const float*)g_, (const float*)be_,
                                     (_Float16*)Y, ldy, (float*)Yf, ldyf,
                                     L[li].KP / 32, flags);
  };

  // ---- stage 0: transpose input -------------------------------------------
  k_transpose<<<256, 256, 0, stream>>>((const float*)d_in[0], pc);

  // ---- SA1 ----------------------------------------------------------------
  k_fps<4096><<<16, 256, 0, stream>>>(pc, fps1, 512);
  k_gather3<<<32, 256, 0, stream>>>(pc, 4096, 512, fps1, xyz1, 16 * 512);
  k_ball_query<<<1024, 256, 0, stream>>>(pc, xyz1, bq1, 512, 32, 0.04f, 4096);
  k_group_sa1<<<1024, 256, 0, stream>>>(pc, xyz1, bq1, G0);
  gemm(G0, 32, 0, d_in[2],  d_in[3],  d_in[4],  G1, 64,  nullptr, 0, 262144, 3);
  gemm(G1, 64, 1, d_in[6],  d_in[7],  d_in[8],  G0, 64,  nullptr, 0, 262144, 3);
  gemm(G0, 64, 2, d_in[10], d_in[11], d_in[12], G1, 128, nullptr, 0, 262144, 3);
  k_maxpool<<<4096, 256, 0, stream>>>(G1, 128, 32, f1, nullptr, 128, 16 * 512 * 128);

  // ---- SA2 ----------------------------------------------------------------
  k_fps<512><<<16, 256, 0, stream>>>(xyz1, fps2, 128);
  k_gather3<<<8, 256, 0, stream>>>(xyz1, 512, 128, fps2, xyz2, 16 * 128);
  k_ball_query<<<256, 256, 0, stream>>>(xyz1, xyz2, bq2, 128, 64, 0.16f, 512);
  k_group_sa2<<<512, 256, 0, stream>>>(xyz1, xyz2, bq2, f1, G0);
  gemm(G0, 160, 3, d_in[14], d_in[15], d_in[16], G1, 128, nullptr, 0, 131072, 3);
  gemm(G1, 128, 4, d_in[18], d_in[19], d_in[20], G0, 128, nullptr, 0, 131072, 3);
  gemm(G0, 128, 5, d_in[22], d_in[23], d_in[24], G1, 256, nullptr, 0, 131072, 3);
  k_maxpool<<<2048, 256, 0, stream>>>(G1, 256, 64, f2, nullptr, 256, 16 * 128 * 256);

  // ---- SA3 (global) -------------------------------------------------------
  k_build_x3<<<8, 256, 0, stream>>>(xyz2, f2, G0);
  gemm(G0, 288, 6, d_in[26], d_in[27], d_in[28], G1, 256,  nullptr, 0, 2048, 3);
  gemm(G1, 256, 7, d_in[30], d_in[31], d_in[32], G0, 512,  nullptr, 0, 2048, 3);
  gemm(G0, 512, 8, d_in[34], d_in[35], d_in[36], G1, 1024, nullptr, 0, 2048, 3);
  k_maxpool<<<64, 256, 0, stream>>>(G1, 1024, 128, nullptr, f3f, 1024, 16 * 1024);

  // ---- FP1 ----------------------------------------------------------------
  k_build_xf1<<<8, 256, 0, stream>>>(f2, f3f, G0);
  gemm(G0, 1280, 9,  d_in[38], d_in[39], d_in[40], G1, 256, nullptr, 0, 2048, 3);
  gemm(G1, 256,  10, d_in[42], d_in[43], d_in[44], f4, 256, nullptr, 0, 2048, 3);

  // ---- FP2 ----------------------------------------------------------------
  k_fp2_build<<<32, 256, 0, stream>>>(xyz1, xyz2, f1, f4, G0);
  gemm(G0, 384, 11, d_in[46], d_in[47], d_in[48], G1, 256, nullptr, 0, 8192, 3);
  gemm(G1, 256, 12, d_in[50], d_in[51], d_in[52], f5, 128, nullptr, 0, 8192, 3);

  // ---- FP3 ----------------------------------------------------------------
  k_fp3_build<<<256, 256, 0, stream>>>(pc, xyz1, f5, G0);
  gemm(G0, 160, 13, d_in[54], d_in[55], d_in[56], G1, 128, nullptr, 0, 65536, 3);
  gemm(G1, 128, 14, d_in[58], d_in[59], d_in[60], G0, 128, nullptr, 0, 65536, 3);
  gemm(G0, 128, 15, d_in[62], d_in[63], d_in[64], G1, 128, nullptr, 0, 65536, 3); // f6

  // ---- head + log_softmax -------------------------------------------------
  gemm(G1, 128, 16, d_in[66], d_in[67], d_in[68], G0, 128, nullptr, 0, 65536, 3); // h
  gemm(G0, 128, 17, b2p, nullptr, nullptr, nullptr, 0, (float*)G1, 64, 65536, 0); // logits
  k_logsm<<<256, 256, 0, stream>>>((const float*)G1, (float*)d_out);
  k_copyf<<<64, 256, 0, stream>>>(f3f, (float*)d_out + (size_t)16 * 50 * 4096, 16 * 1024);
}
